// Attention_79688823210079
// MI455X (gfx1250) — matrix operand order repaired and checked by reading, not compiled, not run on hardware
//
#include <hip/hip_runtime.h>

// ---------------------------------------------------------------------------
// Problem constants (reference: B=4, NQ=1024, NC=8192, D=512, H=8, d=64)
// ---------------------------------------------------------------------------
#define BATCH   4
#define NQ_     1024
#define NC_     8192
#define DMODEL  512
#define HEADS_  8
#define INNER   512   // HEADS_*HDIM

typedef __attribute__((ext_vector_type(16))) __bf16        v16bf;
typedef __attribute__((ext_vector_type(8)))  float         v8f;
typedef __attribute__((ext_vector_type(4)))  unsigned int  v4u;
typedef __attribute__((ext_vector_type(4)))  int           v4i;

union Frag16 { v4u q[2]; unsigned int u[8]; v16bf v; };

#if defined(__has_builtin)
#if __has_builtin(__builtin_amdgcn_global_load_async_to_lds_b128)
#define USE_ASYNC_LDS 1
#endif
#endif

#ifdef USE_ASYNC_LDS
typedef __attribute__((address_space(1))) v4i* as1_v4i;   // global src
typedef __attribute__((address_space(3))) v4i* as3_v4i;   // LDS dst
#endif

__device__ __forceinline__ unsigned short f2bf(float f) {
  union { float f; unsigned int u; } x; x.f = f;
  unsigned int u = x.u;
  return (unsigned short)((u + 0x7FFFu + ((u >> 16) & 1u)) >> 16);  // RNE
}

__device__ __forceinline__ v8f wmma_bf16(const Frag16& a, const Frag16& b, v8f c) {
  return __builtin_amdgcn_wmma_f32_16x16x32_bf16(false, a.v, false, b.v,
                                                 (short)0, c, false, false);
}

// 16-byte global -> LDS copy: async on gfx1250 toolchains that expose the
// builtin (GLOBAL_LOAD_ASYNC_TO_LDS_B128, tracked by ASYNCcnt), else staged.
__device__ __forceinline__ void copy16_g2l(const unsigned int* g, unsigned int* l) {
#ifdef USE_ASYNC_LDS
  __builtin_amdgcn_global_load_async_to_lds_b128((as1_v4i)g, (as3_v4i)l, 0, 0);
#else
  *(v4u*)l = *(const v4u*)g;
#endif
}

// ---------------------------------------------------------------------------
// f32 -> bf16 elementwise convert
// ---------------------------------------------------------------------------
__global__ void k_cvt_bf16(const float* __restrict__ in,
                           unsigned short* __restrict__ out, size_t n) {
  size_t i = (size_t)blockIdx.x * blockDim.x + threadIdx.x;
  if (i < n) out[i] = f2bf(in[i]);
}

// ---------------------------------------------------------------------------
// Weight pack into WMMA B-fragment-major order:
//   Wf[kb][nt][lane][j] (uint = bf16 pair over K) with
//   k2 = kb*16 + (j>>2)*8 + (lane>>4)*4 + (j&3), n = nt*16 + (lane&15)
// so a B fragment is 32 contiguous bytes per lane -> 2 x b128 loads.
// ---------------------------------------------------------------------------
__global__ void k_pack_weight_frag(const float* __restrict__ W,
                                   unsigned int* __restrict__ Wf, int K, int N) {
  size_t i = (size_t)blockIdx.x * blockDim.x + threadIdx.x;
  size_t total = (size_t)(K >> 5) * (N >> 4) * 256;
  if (i >= total) return;
  int j    = (int)(i & 7);
  int lane = (int)((i >> 3) & 31);
  int nt   = (int)((i >> 8) % (N >> 4));
  int kb   = (int)((i >> 8) / (N >> 4));
  int k2 = kb * 16 + (j >> 2) * 8 + (lane >> 4) * 4 + (j & 3);
  int n  = nt * 16 + (lane & 15);
  unsigned int lo = f2bf(W[(size_t)(2 * k2)     * N + n]);
  unsigned int hi = f2bf(W[(size_t)(2 * k2 + 1) * N + n]);
  Wf[i] = lo | (hi << 16);
}

// ---------------------------------------------------------------------------
// bf16 WMMA GEMM, one 16x32 output tile per wave (A fragment reused).
// A: [M][K] bf16 row-major (A fragments are 2 x b128 per lane).
// Wf: fragment-major packed weights. 4 waves / block.
// ---------------------------------------------------------------------------
__global__ __launch_bounds__(128)
void k_gemm_bf16_bfout(const unsigned short* __restrict__ A,
                       const unsigned int*   __restrict__ Wf,
                       unsigned short*       __restrict__ C,
                       int M, int N, int K) {
  const int lane = threadIdx.x & 31;
  const int wave = threadIdx.x >> 5;
  const int tiles2N = N >> 5;
  const int tile  = blockIdx.x * 4 + wave;
  const int mTile = (tile / tiles2N) << 4;
  const int n0    = (tile % tiles2N) << 5;
  if (mTile >= M) return;
  const int half = lane >> 4;
  const int lm   = lane & 15;
  const int K2 = K >> 1, ntiles = N >> 4, nt0 = n0 >> 4;
  const size_t bStride = (size_t)ntiles << 8;

  const unsigned int* aPtr = (const unsigned int*)A + (size_t)(mTile + lm) * K2 + half * 4;
  const unsigned int* bPtr = Wf + ((size_t)nt0 << 8) + (lane << 3);

  v8f c0 = {}, c1 = {};
  for (int kb = 0; kb < (K >> 5); ++kb) {
    Frag16 a, b0, b1;
    a.q[0]  = ((const v4u*)aPtr)[0];  a.q[1]  = ((const v4u*)aPtr)[2];
    b0.q[0] = ((const v4u*)bPtr)[0];  b0.q[1] = ((const v4u*)bPtr)[1];
    b1.q[0] = ((const v4u*)bPtr)[32]; b1.q[1] = ((const v4u*)bPtr)[33];
    __builtin_prefetch(bPtr + bStride, 0, 1);   // speculative: safe past end
    c0 = wmma_bf16(a, b0, c0);
    c1 = wmma_bf16(a, b1, c1);
    aPtr += 16;
    bPtr += bStride;
  }
  // pack adjacent columns across lane pairs -> one dword store per even lane
#pragma unroll
  for (int j = 0; j < 8; ++j) {
    unsigned int w = f2bf(c0[j]);
    unsigned int o = __shfl_xor(w, 1, 32);
    if ((lane & 1) == 0)
      *(unsigned int*)(C + (size_t)(mTile + half * 8 + j) * N + n0 + lm) =
          (w & 0xffffu) | (o << 16);
  }
#pragma unroll
  for (int j = 0; j < 8; ++j) {
    unsigned int w = f2bf(c1[j]);
    unsigned int o = __shfl_xor(w, 1, 32);
    if ((lane & 1) == 0)
      *(unsigned int*)(C + (size_t)(mTile + half * 8 + j) * N + n0 + 16 + lm) =
          (w & 0xffffu) | (o << 16);
  }
}

__global__ __launch_bounds__(128)
void k_gemm_bf16_f32out_bias(const unsigned short* __restrict__ A,
                             const unsigned int*   __restrict__ Wf,
                             const float*          __restrict__ bias,
                             float*                __restrict__ C,
                             int M, int N, int K) {
  const int lane = threadIdx.x & 31;
  const int wave = threadIdx.x >> 5;
  const int tiles2N = N >> 5;
  const int tile  = blockIdx.x * 4 + wave;
  const int mTile = (tile / tiles2N) << 4;
  const int n0    = (tile % tiles2N) << 5;
  if (mTile >= M) return;
  const int half = lane >> 4;
  const int lm   = lane & 15;
  const int K2 = K >> 1, ntiles = N >> 4, nt0 = n0 >> 4;
  const size_t bStride = (size_t)ntiles << 8;

  const unsigned int* aPtr = (const unsigned int*)A + (size_t)(mTile + lm) * K2 + half * 4;
  const unsigned int* bPtr = Wf + ((size_t)nt0 << 8) + (lane << 3);

  v8f c0 = {}, c1 = {};
  for (int kb = 0; kb < (K >> 5); ++kb) {
    Frag16 a, b0, b1;
    a.q[0]  = ((const v4u*)aPtr)[0];  a.q[1]  = ((const v4u*)aPtr)[2];
    b0.q[0] = ((const v4u*)bPtr)[0];  b0.q[1] = ((const v4u*)bPtr)[1];
    b1.q[0] = ((const v4u*)bPtr)[32]; b1.q[1] = ((const v4u*)bPtr)[33];
    __builtin_prefetch(bPtr + bStride, 0, 1);
    c0 = wmma_bf16(a, b0, c0);
    c1 = wmma_bf16(a, b1, c1);
    aPtr += 16;
    bPtr += bStride;
  }
  const float bb0 = bias[n0 + lm], bb1 = bias[n0 + 16 + lm];
#pragma unroll
  for (int j = 0; j < 8; ++j) {
    size_t row = (size_t)(mTile + half * 8 + j) * N;
    C[row + n0 + lm]      = c0[j] + bb0;
    C[row + n0 + 16 + lm] = c1[j] + bb1;
  }
}

// ---------------------------------------------------------------------------
// Rearrange KV (bf16 [b][nc][1024], K cols 0..511, V cols 512..1023) into
// WMMA-fragment-major buffers:
//  Ktf[bh][cTile(16 ctx)][dslice(0/1)][lane*8+j] : pair over head-dim
//       value for (ctx = cTile*16 + lane&15, d2 = dslice*16 + (j>>2)*8 + (lane>>4)*4 + (j&3))
//  Vpf[bh][cChunk(32 ctx)][nd(0..3)][lane*8+j]  : pair over ctx
//       value for (j2 = cChunk*16 + (j>>2)*8 + (lane>>4)*4 + (j&3), d = nd*16 + lane&15)
// Each 32-ctx chunk is a contiguous 4 KiB block in both buffers.
// ---------------------------------------------------------------------------
__global__ void k_build_ktf(const unsigned short* __restrict__ KV,
                            unsigned int* __restrict__ Ktf) {
  size_t i = (size_t)blockIdx.x * blockDim.x + threadIdx.x;
  const size_t total = (size_t)BATCH * HEADS_ * (NC_ / 16) * 2 * 256;
  if (i >= total) return;
  int j    = (int)(i & 7);
  int lane = (int)((i >> 3) & 31);
  int ds   = (int)((i >> 8) & 1);
  int ct   = (int)((i >> 9) & (NC_ / 16 - 1));
  int bh   = (int)(i >> 18);
  int b = bh >> 3, h = bh & 7;
  int ctx = ct * 16 + (lane & 15);
  int d2  = ds * 16 + (j >> 2) * 8 + (lane >> 4) * 4 + (j & 3);
  const unsigned int* KV32 = (const unsigned int*)KV;
  Ktf[i] = KV32[(size_t)(b * NC_ + ctx) * 512 + h * 32 + d2];
}

__global__ void k_build_vpf(const unsigned short* __restrict__ KV,
                            unsigned int* __restrict__ Vpf) {
  size_t i = (size_t)blockIdx.x * blockDim.x + threadIdx.x;
  const size_t total = (size_t)BATCH * HEADS_ * (NC_ / 32) * 4 * 256;
  if (i >= total) return;
  int j    = (int)(i & 7);
  int lane = (int)((i >> 3) & 31);
  int nd   = (int)((i >> 8) & 3);
  int ch   = (int)((i >> 10) & (NC_ / 32 - 1));
  int bh   = (int)(i >> 18);
  int b = bh >> 3, h = bh & 7;
  int j2 = ch * 16 + (j >> 2) * 8 + (lane >> 4) * 4 + (j & 3);
  int d  = nd * 16 + (lane & 15);
  unsigned int lo = KV[(size_t)(b * NC_ + 2 * j2)     * 1024 + 512 + h * 64 + d];
  unsigned int hi = KV[(size_t)(b * NC_ + 2 * j2 + 1) * 1024 + 512 + h * 64 + d];
  Vpf[i] = lo | (hi << 16);
}

// ---------------------------------------------------------------------------
// Flash attention. 4 waves/block share (b,h); each wave owns a 16-query strip.
// Per 32-ctx chunk: workgroup async-copies the 8 KiB fragment block (4 KiB K +
// 4 KiB V) into a double-buffered LDS stage, then each wave does
// 16 x ds_load_b128 -> 8 x v_wmma_f32_16x16x32_bf16 + f32 online softmax.
// ---------------------------------------------------------------------------
__global__ __launch_bounds__(128)
void k_flash_attn(const unsigned short* __restrict__ Qbf,   // [b][nq][512]
                  const unsigned int*   __restrict__ Ktf,
                  const unsigned int*   __restrict__ Vpf,
                  unsigned short*       __restrict__ Obf) { // [b][nq][512]
  __shared__ alignas(16) unsigned int ldsKV[2][2048];       // 2 x (1024 K + 1024 V)
  __shared__ unsigned short Pl[4][16 * 32];                 // per-wave P tile
  const int tid  = threadIdx.x;
  const int lane = tid & 31;
  const int wave = tid >> 5;
  const int tilesQ = NQ_ / 64;
  const int bh = blockIdx.x / tilesQ;
  const int q0 = (blockIdx.x % tilesQ) * 64 + wave * 16;
  const int b = bh >> 3, h = bh & 7;
  const int half = lane >> 4;
  const int lm   = lane & 15;

  const unsigned int* KtH = Ktf + (size_t)bh * (NC_ / 16) * 512;
  const unsigned int* VpH = Vpf + (size_t)bh * (NC_ / 32) * 1024;

  // Q A-fragments (2 x b128 each), head-dim slices 0..31 / 32..63
  Frag16 qa0, qa1;
  {
    const v4u* qp = (const v4u*)((const unsigned int*)Qbf +
                                 (size_t)(b * NQ_ + q0 + lm) * 256 + h * 32 + half * 4);
    qa0.q[0] = qp[0]; qa0.q[1] = qp[2];
    qa1.q[0] = qp[4]; qa1.q[1] = qp[6];
  }

  v8f o0 = {}, o1 = {}, o2 = {}, o3 = {};
  float m_i[8], l_i[8];
#pragma unroll
  for (int j = 0; j < 8; ++j) { m_i[j] = -1e30f; l_i[j] = 0.0f; }

  auto issue_copy = [&](int buf, int c0) {
    const unsigned int* gK = KtH + (size_t)(c0 >> 4) * 512;
    const unsigned int* gV = VpH + (size_t)(c0 >> 5) * 1024;
    unsigned int* lb = &ldsKV[buf][0];
#pragma unroll
    for (int r = 0; r < 2; ++r) {
      int seg = (tid + r * 128) << 2;          // dword offset of a 16B segment
      copy16_g2l(gK + seg, lb + seg);
      copy16_g2l(gV + seg, lb + 1024 + seg);
    }
  };

  const float scale = 0.125f;  // 64^-0.5
  issue_copy(0, 0);

  for (int c0 = 0; c0 < NC_; c0 += 32) {
    const int buf = (c0 >> 5) & 1;
    const bool hasNext = (c0 + 32) < NC_;
    if (hasNext) issue_copy(buf ^ 1, c0 + 32);
#ifdef USE_ASYNC_LDS
    if (hasNext) asm volatile("s_wait_asynccnt 0x4" ::: "memory");
    else         asm volatile("s_wait_asynccnt 0x0" ::: "memory");
#endif
    __syncthreads();   // chunk data visible to all waves

    const unsigned int* lk = &ldsKV[buf][0];
    Frag16 kb00, kb01, kb10, kb11;
    { const v4u* p = (const v4u*)(lk +   0 + (lane << 3)); kb00.q[0] = p[0]; kb00.q[1] = p[1]; }
    { const v4u* p = (const v4u*)(lk + 256 + (lane << 3)); kb01.q[0] = p[0]; kb01.q[1] = p[1]; }
    { const v4u* p = (const v4u*)(lk + 512 + (lane << 3)); kb10.q[0] = p[0]; kb10.q[1] = p[1]; }
    { const v4u* p = (const v4u*)(lk + 768 + (lane << 3)); kb11.q[0] = p[0]; kb11.q[1] = p[1]; }

    v8f s0 = {}, s1 = {};
    s0 = wmma_bf16(qa0, kb00, s0);
    s0 = wmma_bf16(qa1, kb01, s0);
    s1 = wmma_bf16(qa0, kb10, s1);
    s1 = wmma_bf16(qa1, kb11, s1);

    // f32 online softmax; row r = half*8+j lives in VGPR j across 16 lanes
#pragma unroll
    for (int j = 0; j < 8; ++j) { s0[j] *= scale; s1[j] *= scale; }
#pragma unroll
    for (int j = 0; j < 8; ++j) {
      float v = fmaxf(s0[j], s1[j]);
      v = fmaxf(v, __shfl_xor(v, 1, 32));
      v = fmaxf(v, __shfl_xor(v, 2, 32));
      v = fmaxf(v, __shfl_xor(v, 4, 32));
      v = fmaxf(v, __shfl_xor(v, 8, 32));
      float mN = fmaxf(m_i[j], v);
      float al = __expf(m_i[j] - mN);
      float p0 = __expf(s0[j] - mN);
      float p1 = __expf(s1[j] - mN);
      float rs = p0 + p1;
      rs += __shfl_xor(rs, 1, 32);
      rs += __shfl_xor(rs, 2, 32);
      rs += __shfl_xor(rs, 4, 32);
      rs += __shfl_xor(rs, 8, 32);
      l_i[j] = l_i[j] * al + rs;
      m_i[j] = mN;
      o0[j] *= al; o1[j] *= al; o2[j] *= al; o3[j] *= al;
      Pl[wave][(half * 8 + j) * 32 + lm]      = f2bf(p0);
      Pl[wave][(half * 8 + j) * 32 + 16 + lm] = f2bf(p1);
    }

    // reload P as A-fragment (C-layout -> A-layout via per-wave LDS tile)
    Frag16 pa;
#pragma unroll
    for (int j = 0; j < 8; ++j) {
      int kp = ((j >= 4) ? 16 : 0) + half * 8 + ((j & 3) << 1);
      unsigned int lo = Pl[wave][lm * 32 + kp];
      unsigned int hi = Pl[wave][lm * 32 + kp + 1];
      pa.u[j] = lo | (hi << 16);
    }

    Frag16 vb0, vb1, vb2, vb3;
    { const v4u* p = (const v4u*)(lk + 1024 + (lane << 3)); vb0.q[0] = p[0]; vb0.q[1] = p[1]; }
    { const v4u* p = (const v4u*)(lk + 1280 + (lane << 3)); vb1.q[0] = p[0]; vb1.q[1] = p[1]; }
    { const v4u* p = (const v4u*)(lk + 1536 + (lane << 3)); vb2.q[0] = p[0]; vb2.q[1] = p[1]; }
    { const v4u* p = (const v4u*)(lk + 1792 + (lane << 3)); vb3.q[0] = p[0]; vb3.q[1] = p[1]; }
    o0 = wmma_bf16(pa, vb0, o0);
    o1 = wmma_bf16(pa, vb1, o1);
    o2 = wmma_bf16(pa, vb2, o2);
    o3 = wmma_bf16(pa, vb3, o3);

    __syncthreads();   // all reads of buf done before it is refilled
  }

  // normalize + write attention output bf16 [b][nq][h*64+d]
#pragma unroll
  for (int j = 0; j < 8; ++j) {
    float inv = 1.0f / l_i[j];
    size_t base = (size_t)(b * NQ_ + q0 + half * 8 + j) * 512 + h * 64;
    Obf[base +      lm] = f2bf(o0[j] * inv);
    Obf[base + 16 + lm] = f2bf(o1[j] * inv);
    Obf[base + 32 + lm] = f2bf(o2[j] * inv);
    Obf[base + 48 + lm] = f2bf(o3[j] * inv);
  }
}

// ---------------------------------------------------------------------------
// Host-side orchestration
// ---------------------------------------------------------------------------
extern "C" void kernel_launch(void* const* d_in, const int* in_sizes, int n_in,
                              void* d_out, int out_size, void* d_ws, size_t ws_size,
                              hipStream_t stream) {
  const float* x    = (const float*)d_in[0];
  const float* ctx  = (const float*)d_in[1];
  const float* Wq   = (const float*)d_in[2];
  const float* Wkv  = (const float*)d_in[3];
  const float* Wout = (const float*)d_in[4];
  const float* bout = (const float*)d_in[5];
  float* out = (float*)d_out;

  const size_t xe = (size_t)BATCH * NQ_ * DMODEL;   // 2,097,152
  const size_t ce = (size_t)BATCH * NC_ * DMODEL;   // 16,777,216
  const int Mq = BATCH * NQ_;                       // 4096
  const int Mc = BATCH * NC_;                       // 32768

  char* ws = (char*)d_ws;
  size_t off = 0;
  auto alloc = [&](size_t bytes) -> void* {
    off = (off + 255) & ~(size_t)255;
    void* p = ws + off;
    off += bytes;
    return p;
  };
  unsigned short* x_bf   = (unsigned short*)alloc(xe * 2);
  unsigned short* ctx_bf = (unsigned short*)alloc(ce * 2);
  unsigned int*   Wq_f   = (unsigned int*)  alloc((size_t)256 * 512  * 4);
  unsigned int*   Wkv_f  = (unsigned int*)  alloc((size_t)256 * 1024 * 4);
  unsigned int*   Wout_f = (unsigned int*)  alloc((size_t)256 * 512  * 4);
  unsigned short* Q_bf   = (unsigned short*)alloc((size_t)Mq * INNER * 2);
  unsigned short* KV_bf  = (unsigned short*)alloc((size_t)Mc * 1024 * 2);
  unsigned int*   Ktf    = (unsigned int*)  alloc((size_t)BATCH * HEADS_ * (NC_/16) * 512 * 4);
  unsigned int*   Vpf    = (unsigned int*)  alloc((size_t)BATCH * HEADS_ * (NC_/32) * 1024 * 4);
  unsigned short* aO_bf  = (unsigned short*)alloc((size_t)Mq * INNER * 2);
  (void)ws_size; (void)in_sizes; (void)n_in; (void)out_size;

  // 1) convert activations, pack weights fragment-major
  k_cvt_bf16<<<(int)((xe + 255) / 256), 256, 0, stream>>>(x,   x_bf,   xe);
  k_cvt_bf16<<<(int)((ce + 255) / 256), 256, 0, stream>>>(ctx, ctx_bf, ce);
  k_pack_weight_frag<<<512,  256, 0, stream>>>(Wq,   Wq_f,   512, 512);
  k_pack_weight_frag<<<1024, 256, 0, stream>>>(Wkv,  Wkv_f,  512, 1024);
  k_pack_weight_frag<<<512,  256, 0, stream>>>(Wout, Wout_f, 512, 512);

  // 2) Q = x @ Wq
  k_gemm_bf16_bfout<<<(Mq / 16) * (INNER / 32) / 4, 128, 0, stream>>>(
      x_bf, Wq_f, Q_bf, Mq, INNER, DMODEL);

  // 3) KV = context @ Wkv
  k_gemm_bf16_bfout<<<(Mc / 16) * (1024 / 32) / 4, 128, 0, stream>>>(
      ctx_bf, Wkv_f, KV_bf, Mc, 1024, DMODEL);

  // 4) swizzle K^T / V into WMMA fragment-major chunk blocks
  k_build_ktf<<<32768, 256, 0, stream>>>(KV_bf, Ktf);
  k_build_vpf<<<32768, 256, 0, stream>>>(KV_bf, Vpf);

  // 5) flash attention
  k_flash_attn<<<BATCH * HEADS_ * (NQ_ / 64), 128, 0, stream>>>(Q_bf, Ktf, Vpf, aO_bf);

  // 6) out = attnO @ Wout + bout
  k_gemm_bf16_f32out_bias<<<(Mq / 16) * (DMODEL / 32) / 4, 128, 0, stream>>>(
      aO_bf, Wout_f, bout, out, Mq, DMODEL, INNER);
}
